// Decoder_61701500174895
// MI455X (gfx1250) — matrix operand order repaired and checked by reading, not compiled, not run on hardware
//
#include <hip/hip_runtime.h>
#include <stdint.h>

// ---------------------------------------------------------------------------
// Detection post-processing (decode + per-class NMS + top-200) for MI455X.
// B=1, N=2048 anchors, L=21 classes. Latency-bound: design keeps each class
// fully resident in one WGP's LDS, 21 workgroups concurrent.
// CDNA5 paths: v_wmma_f32_16x16x32_f16 for exact 0/1 prefix scan (keep-cap),
// global_load_async_to_lds_b128 gather (guarded), global_prefetch_b8.
// ---------------------------------------------------------------------------

#define L_CLS 21
#define N_BOX 2048
#define MAX_TOTAL 200
#define NMS_THR 0.5f
#define SCORE_THR 0.5f

typedef _Float16 v16h __attribute__((ext_vector_type(16)));
typedef float    v8f  __attribute__((ext_vector_type(8)));

#if defined(__HIP_DEVICE_COMPILE__) && \
    __has_builtin(__builtin_amdgcn_global_load_async_to_lds_b128) && \
    __has_builtin(__builtin_amdgcn_s_wait_asynccnt)
#define USE_ASYNC_LDS 1
#else
#define USE_ASYNC_LDS 0
#endif

#if USE_ASYNC_LDS
// Builtin signature (from hipcc diagnostic): param0 is
// 'int __attribute__((vector_size(16))) addrspace(1)*' (printed "__device__ *").
typedef int v4i_b128 __attribute__((vector_size(16)));
typedef __attribute__((address_space(1))) v4i_b128 as1_v4i;
typedef __attribute__((address_space(3))) v4i_b128 as3_v4i;

__device__ __forceinline__ void async_copy_b128_to_lds(const float* gsrc, float* lds_dst) {
  // Global->LDS async DMA (GLOBAL_LOAD_ASYNC_TO_LDS_B128, tracked by ASYNCcnt).
  // LDS generic pointers carry the LDS byte offset in the low 32 bits.
  __builtin_amdgcn_global_load_async_to_lds_b128(
      (as1_v4i*)(uintptr_t)gsrc,
      (as3_v4i*)(uint32_t)(uintptr_t)lds_dst,
      0, 0);
}
#endif

// ---------------------------------------------------------------------------
// Kernel 1: decode boxes + argmax score masking, write class-major layouts.
//   boxes_cls  [L][N][4], scores_cls [L][N]
// ---------------------------------------------------------------------------
__global__ void decode_kernel(const float* __restrict__ rois,
                              const float* __restrict__ deltas,
                              const float* __restrict__ probs,
                              float* __restrict__ boxes_cls,
                              float* __restrict__ scores_cls) {
  int n = blockIdx.x * blockDim.x + threadIdx.x;
  if (n >= N_BOX) return;

  float r0 = rois[n * 4 + 0], r1 = rois[n * 4 + 1];
  float r2 = rois[n * 4 + 2], r3 = rois[n * 4 + 3];
  float h = r2 - r0, w = r3 - r1;
  float cy = r0 + 0.5f * h, cx = r1 + 0.5f * w;

  // argmax (first max wins, matching jnp.argmax)
  const float* p = probs + n * L_CLS;
  int am = 0;
  float best = p[0];
#pragma unroll
  for (int c = 1; c < L_CLS; ++c) {
    float v = p[c];
    if (v > best) { best = v; am = c; }
  }
  float mask = (am != 0) ? 1.0f : 0.0f;

  const float* d = deltas + n * (L_CLS * 4);
#pragma unroll 3
  for (int c = 0; c < L_CLS; ++c) {
    float dy = d[c * 4 + 0] * 0.1f;
    float dx = d[c * 4 + 1] * 0.1f;
    float dh = d[c * 4 + 2] * 0.2f;
    float dw = d[c * 4 + 3] * 0.2f;
    float nh = __expf(dh) * h;   // f32 exp
    float nw = __expf(dw) * w;
    float ncy = dy * h + cy;
    float ncx = dx * w + cx;
    int o = c * N_BOX + n;
    boxes_cls[o * 4 + 0] = ncy - 0.5f * nh;
    boxes_cls[o * 4 + 1] = ncx - 0.5f * nw;
    boxes_cls[o * 4 + 2] = ncy + 0.5f * nh;
    boxes_cls[o * 4 + 3] = ncx + 0.5f * nw;
    scores_cls[o] = mask * p[c];
  }
}

// ---------------------------------------------------------------------------
// Kernel 2: one workgroup per class. Bitonic sort (desc score, asc idx),
// async gather of boxes to LDS, greedy NMS, WMMA-based keep cumsum cap.
// ---------------------------------------------------------------------------
__global__ __launch_bounds__(1024) void nms_kernel(
    const float* __restrict__ boxes_cls,
    const float* __restrict__ scores_cls,
    float* __restrict__ boxes_sorted,
    float* __restrict__ scores_sorted,
    int* __restrict__ keep_out) {
  const int c = blockIdx.x;
  const int tid = threadIdx.x;
  const int nth = blockDim.x;

  __shared__ float               s_sc[N_BOX];
  __shared__ int                 s_id[N_BOX];      // reused as f32 cumsum later
  __shared__ __align__(16) float s_bx[N_BOX * 4];
  __shared__ int                 s_keep[N_BOX];
  __shared__ float               s_off[N_BOX / 16];

  // ---- load scores + init ids; prefetch this class's box rows toward WGP ----
  for (int t = tid; t < N_BOX; t += nth) {
    s_sc[t] = scores_cls[c * N_BOX + t];
    s_id[t] = t;
    __builtin_prefetch(&boxes_cls[(size_t)(c * N_BOX + t) * 4], 0, 3);
  }

  // ---- bitonic sort: descending score, ties -> ascending index (stable) ----
  for (unsigned k = 2; k <= N_BOX; k <<= 1) {
    for (unsigned j = k >> 1; j > 0; j >>= 1) {
      __syncthreads();
      for (unsigned idx = tid; idx < N_BOX; idx += nth) {
        unsigned ixj = idx ^ j;
        if (ixj > idx) {
          float sa = s_sc[idx], sb = s_sc[ixj];
          int ia = s_id[idx], ib = s_id[ixj];
          bool up = ((idx & k) == 0);
          // before(b,a): b strictly precedes a in final (descending) order
          bool b_before_a = (sb > sa) || (sb == sa && ib < ia);
          if (up == b_before_a) {
            s_sc[idx] = sb; s_sc[ixj] = sa;
            s_id[idx] = ib; s_id[ixj] = ia;
          }
        }
      }
    }
  }
  __syncthreads();

  // ---- gather sorted boxes into LDS (async DMA path if available) ----
  for (int t = tid; t < N_BOX; t += nth) {
    int si = s_id[t];
    const float* src = &boxes_cls[(size_t)(c * N_BOX + si) * 4];
#if USE_ASYNC_LDS
    async_copy_b128_to_lds(src, &s_bx[t * 4]);
#else
    float4 v = *reinterpret_cast<const float4*>(src);
    *reinterpret_cast<float4*>(&s_bx[t * 4]) = v;
#endif
    s_keep[t] = 1;
  }
#if USE_ASYNC_LDS
  __builtin_amdgcn_s_wait_asynccnt(0);
#endif
  __syncthreads();

  // ---- greedy NMS: serial over pivots, parallel over victims ----
  for (int i = 0; i < N_BOX - 1; ++i) {
    __syncthreads();
    if (s_keep[i]) {                      // uniform across workgroup
      float y1 = s_bx[i * 4 + 0], x1 = s_bx[i * 4 + 1];
      float y2 = s_bx[i * 4 + 2], x2 = s_bx[i * 4 + 3];
      float ai = (y2 - y1) * (x2 - x1);
      for (int jj = i + 1 + tid; jj < N_BOX; jj += nth) {
        if (s_keep[jj]) {
          float by1 = s_bx[jj * 4 + 0], bx1 = s_bx[jj * 4 + 1];
          float by2 = s_bx[jj * 4 + 2], bx2 = s_bx[jj * 4 + 3];
          float aj = (by2 - by1) * (bx2 - bx1);
          float ih = fmaxf(fminf(y2, by2) - fmaxf(y1, by1), 0.0f);
          float iw = fmaxf(fminf(x2, bx2) - fmaxf(x1, bx1), 0.0f);
          float inter = ih * iw;
          float iou = inter / (ai + aj - inter + 1e-9f);
          if (iou > NMS_THR) s_keep[jj] = 0;
        }
      }
    }
  }
  __syncthreads();

  // ---- keep-cap via WMMA segmented inclusive scan of 0/1 flags ----
  // A (16x32 f16): A[m][k] = (k<=m && k<16). B (32x16 f16): B[k][n] =
  // keep[tile*256 + n*16 + k] (k<16).  C[m][n] = inclusive count at element
  // tile*256 + n*16 + m.  Exact: 0/1 in f16, f32 accumulate.
  float* s_cum = reinterpret_cast<float*>(s_id);  // s_id dead after gather
  if (tid < 32) {
    int lane = tid;
    v16h A = {};
    if (lane < 16) {
#pragma unroll
      for (int e = 0; e < 8; ++e)               // K = e (ISA A f16 layout)
        A[e] = (e <= lane) ? (_Float16)1 : (_Float16)0;
    } else {
      int m = lane - 16;
#pragma unroll
      for (int e = 0; e < 8; ++e)               // K = 8+e
        A[e] = ((8 + e) <= m) ? (_Float16)1 : (_Float16)0;
    }
    for (int t = 0; t < N_BOX / 256; ++t) {
      v16h B = {};
      {
        int base = t * 256 + (lane & 15) * 16 + ((lane < 16) ? 0 : 8);
#pragma unroll
        for (int e = 0; e < 8; ++e)             // K = 8*(lane/16)+e, N = lane%16
          B[e] = (_Float16)(float)s_keep[base + e];
      }
      v8f C = {};
      C = __builtin_amdgcn_wmma_f32_16x16x32_f16(
          /*neg_a=*/false, A, /*neg_b=*/false, B,
          /*c_mod=*/(short)0, C, /*reuse_a=*/false, /*reuse_b=*/false);
      int n16 = lane & 15;
      int mbase = t * 256 + n16 * 16 + ((lane < 16) ? 0 : 8);
#pragma unroll
      for (int p = 0; p < 8; ++p) s_cum[mbase + p] = C[p];
    }
  }
  __syncthreads();
  if (tid == 0) {                               // 128 block offsets (exclusive)
    float run = 0.0f;
    for (int b = 0; b < N_BOX / 16; ++b) {
      s_off[b] = run;
      run += s_cum[b * 16 + 15];
    }
  }
  __syncthreads();
  for (int t = tid; t < N_BOX; t += nth) {
    float total = s_cum[t] + s_off[t >> 4];     // inclusive cumsum of keep
    if (total > (float)MAX_TOTAL) s_keep[t] = 0;
  }
  __syncthreads();

  // ---- write sorted boxes / scores / capped keep ----
  for (int t = tid; t < N_BOX; t += nth) {
    int o = c * N_BOX + t;
    scores_sorted[o] = s_sc[t];
    keep_out[o] = s_keep[t];
    boxes_sorted[o * 4 + 0] = s_bx[t * 4 + 0];
    boxes_sorted[o * 4 + 1] = s_bx[t * 4 + 1];
    boxes_sorted[o * 4 + 2] = s_bx[t * 4 + 2];
    boxes_sorted[o * 4 + 3] = s_bx[t * 4 + 3];
  }
}

// ---------------------------------------------------------------------------
// Kernel 3: 21-way merge of per-class sorted kept lists -> top-200 (matches
// lax.top_k stability: lower flat index wins ties), plus fallback slot.
// Output layout: boxes[200][4] @0, labels[200] @800, scores[200] @1000.
// ---------------------------------------------------------------------------
__global__ void select_kernel(const float* __restrict__ boxes_sorted,
                              const float* __restrict__ scores_sorted,
                              const int* __restrict__ keep,
                              float* __restrict__ out) {
  int tid = threadIdx.x;
  for (int i = tid; i < MAX_TOTAL * 6; i += blockDim.x) out[i] = 0.0f;
  __syncthreads();
  if (tid != 0) return;

  // advance to next entry with keep && score > thr (thr<0 => keep only)
  auto adv = [&](int cls, int t, float thr) {
    while (t < N_BOX) {
      int o = cls * N_BOX + t;
      if (keep[o] && scores_sorted[o] > thr) break;
      ++t;
    }
    return t;
  };

  int p[L_CLS];
  for (int cl = 0; cl < L_CLS; ++cl) p[cl] = adv(cl, 0, SCORE_THR);

  int cnt = 0;
  for (int slot = 0; slot < MAX_TOTAL; ++slot) {
    int bc = -1;
    float bs = -1.0f;
    for (int cl = 0; cl < L_CLS; ++cl) {        // ascending class = tie-break
      if (p[cl] < N_BOX) {
        float s = scores_sorted[cl * N_BOX + p[cl]];
        if (s > bs) { bs = s; bc = cl; }
      }
    }
    if (bc < 0) break;
    int src = bc * N_BOX + p[bc];
    out[slot * 4 + 0] = fminf(fmaxf(boxes_sorted[src * 4 + 0], 0.0f), 1.0f);
    out[slot * 4 + 1] = fminf(fmaxf(boxes_sorted[src * 4 + 1], 0.0f), 1.0f);
    out[slot * 4 + 2] = fminf(fmaxf(boxes_sorted[src * 4 + 2], 0.0f), 1.0f);
    out[slot * 4 + 3] = fminf(fmaxf(boxes_sorted[src * 4 + 3], 0.0f), 1.0f);
    out[MAX_TOTAL * 4 + slot] = (float)bc;
    out[MAX_TOTAL * 5 + slot] = bs;
    ++cnt;
    p[bc] = adv(bc, p[bc] + 1, SCORE_THR);
  }

  if (cnt == 0) {
    // fallback: best kept (unthresholded) entry across all classes
    int bc = -1, bp = 0;
    float bs = -1.0f;
    for (int cl = 0; cl < L_CLS; ++cl) {
      int t = adv(cl, 0, -1.0f);                // keep-only
      if (t < N_BOX) {
        float s = scores_sorted[cl * N_BOX + t];
        if (s > bs) { bs = s; bc = cl; bp = t; }
      }
    }
    if (bc >= 0 && bs >= 0.001f) {
      int src = bc * N_BOX + bp;
      out[0] = fminf(fmaxf(boxes_sorted[src * 4 + 0], 0.0f), 1.0f);
      out[1] = fminf(fmaxf(boxes_sorted[src * 4 + 1], 0.0f), 1.0f);
      out[2] = fminf(fmaxf(boxes_sorted[src * 4 + 2], 0.0f), 1.0f);
      out[3] = fminf(fmaxf(boxes_sorted[src * 4 + 3], 0.0f), 1.0f);
      out[MAX_TOTAL * 4 + 0] = (float)bc;
      out[MAX_TOTAL * 5 + 0] = bs;
    }
  }
}

// ---------------------------------------------------------------------------
extern "C" void kernel_launch(void* const* d_in, const int* in_sizes, int n_in,
                              void* d_out, int out_size, void* d_ws, size_t ws_size,
                              hipStream_t stream) {
  const float* rois   = (const float*)d_in[0];  // (1,2048,4)
  const float* deltas = (const float*)d_in[1];  // (1,2048,84)
  const float* probs  = (const float*)d_in[2];  // (1,2048,21)

  float* ws = (float*)d_ws;
  float* boxes_cls     = ws;                                   // 21*2048*4
  float* scores_cls    = ws + (size_t)L_CLS * N_BOX * 4;       // 21*2048
  float* boxes_sorted  = scores_cls + (size_t)L_CLS * N_BOX;   // 21*2048*4
  float* scores_sorted = boxes_sorted + (size_t)L_CLS * N_BOX * 4;
  int*   keep          = (int*)(scores_sorted + (size_t)L_CLS * N_BOX);

  decode_kernel<<<N_BOX / 256, 256, 0, stream>>>(rois, deltas, probs,
                                                 boxes_cls, scores_cls);
  nms_kernel<<<L_CLS, 1024, 0, stream>>>(boxes_cls, scores_cls,
                                         boxes_sorted, scores_sorted, keep);
  select_kernel<<<1, 256, 0, stream>>>(boxes_sorted, scores_sorted, keep,
                                       (float*)d_out);
}
